// MultiHeadSelfAttention_76124000354897
// MI455X (gfx1250) — matrix operand order repaired and checked by reading, not compiled, not run on hardware
//
#include <hip/hip_runtime.h>

// ---------------------------------------------------------------------------
// MI455X (gfx1250) multi-head self-attention.
// bf16 WMMA (f32 accum), double-buffered LDS pipeline:
//   A tiles via TDM tensor_load_to_lds (D# padding -> 40-bf16 LDS row stride)
//   B tiles via per-lane GLOBAL_LOAD_ASYNC_TO_LDS_B128
//   next tile's DMA overlaps current tile's WMMAs (ASYNCcnt/TENSORcnt <= N)
// B=4, S=1024, d_model=512, H=8 heads of width 512 each.
// ---------------------------------------------------------------------------

typedef __attribute__((ext_vector_type(16))) __bf16 v16bf;
typedef __attribute__((ext_vector_type(8)))  __bf16 v8bf;
typedef __attribute__((ext_vector_type(8)))  float  v8f;
typedef unsigned int v4u __attribute__((ext_vector_type(4)));
typedef int          v4i __attribute__((ext_vector_type(4)));
typedef int          v8i __attribute__((ext_vector_type(8)));

constexpr int B_  = 4;
constexpr int S_  = 1024;
constexpr int D_  = 512;
constexpr int H_  = 8;
constexpr int HD_ = H_ * D_;   // 4096
constexpr int M_  = B_ * S_;   // 4096

// GEMM tiling: 128x128 block tile, 8 waves of 64x32, K-step 32, 2 LDS buffers.
constexpr int BM  = 128;
constexpr int BN  = 128;
constexpr int BK  = 32;
constexpr int PAD = 8;          // 16B row padding -> conflict-free ds_load_b128
constexpr int LDT = BK + PAD;   // 40 bf16 = 80B LDS row stride

// ---------------------------------------------------------------------------
// CDNA5 async global->LDS copy (ASYNCcnt-tracked). The offset: immediate is
// added to BOTH the LDS and global addresses, so one base pair covers 32B.
// ---------------------------------------------------------------------------
__device__ __forceinline__ void async_copy32B(void* lds, const void* gaddr) {
  const unsigned ldsOff = (unsigned)(unsigned long long)lds;  // low 32 bits = LDS byte addr
  asm volatile("global_load_async_to_lds_b128 %0, %1, off\n\t"
               "global_load_async_to_lds_b128 %0, %1, off offset:16"
               :: "v"(ldsOff), "v"(gaddr) : "memory");
}

__device__ __forceinline__ void wait_async_le0() {
  asm volatile("s_wait_asynccnt 0x0" ::: "memory");
}
__device__ __forceinline__ void wait_async_le2() {
  asm volatile("s_wait_asynccnt 0x2" ::: "memory");
}

// ---------------------------------------------------------------------------
// TDM: load a BM x BK bf16 tile (row stride ldElems) into LDS with padding
// (pad_interval=3 -> pad after every 16 DWORDs, pad_amount=3 -> 4 DWORDs) so
// the LDS image has exactly the LDT=40 bf16 row stride the fragments expect.
// D# bit packing per CDNA5 ISA 8.3/8.4. Issued by one wave; TENSORcnt-tracked.
// ---------------------------------------------------------------------------
__device__ __forceinline__ void tdm_load_tile_a(const __bf16* gsrc, void* ldsDst,
                                                int ldElems) {
  const unsigned long long ga = (unsigned long long)gsrc;
  const unsigned lds = (unsigned)(unsigned long long)ldsDst;
  v4u g0;
  g0[0] = 1u;                                                // count=1, user mode
  g0[1] = lds;                                               // lds_addr
  g0[2] = (unsigned)(ga & 0xFFFFFFFFu);                      // global_addr[31:0]
  g0[3] = (unsigned)((ga >> 32) & 0x01FFFFFFu) | (2u << 30); // global_addr[56:32], type=2
  v8i g1;
  g1[0] = (int)((1u << 16)      // data_size = 2B
              | (1u << 20)      // pad_enable
              | (3u << 22)      // pad_interval: 16 DWORDs between pads
              | (3u << 25));    // pad_amount: 4 DWORDs (8 bf16)
  g1[1] = (int)((unsigned)BK << 16);        // tensor_dim0[15:0]=32
  g1[2] = (int)((unsigned)BM << 16);        // tensor_dim0 hi=0 | tensor_dim1[15:0]=128
  g1[3] = (int)((unsigned)BK << 16);        // tensor_dim1 hi=0 | tile_dim0=32
  g1[4] = (int)(unsigned)BM;                // tile_dim1=128, tile_dim2=0 (2-D tile)
  g1[5] = ldElems;                          // tensor_dim0_stride[31:0] (elements)
  g1[6] = 0;                                // stride hi, tensor_dim1_stride lo (unused)
  g1[7] = 0;
  const v4i gz = {0, 0, 0, 0};
#if __clang_major__ >= 23
  const v8i gz8 = {0, 0, 0, 0, 0, 0, 0, 0};
  __builtin_amdgcn_tensor_load_to_lds(g0, g1, gz, gz, gz8, 0);
#else
  __builtin_amdgcn_tensor_load_to_lds(g0, g1, gz, gz, 0);
#endif
}

// ---------------------------------------------------------------------------
// f32 -> bf16 elementwise convert (for x)
// ---------------------------------------------------------------------------
__global__ __launch_bounds__(256) void cvt_f32_bf16(const float* __restrict__ in,
                                                    __bf16* __restrict__ out, int n) {
  int i = blockIdx.x * 256 + threadIdx.x;
  const int stride = gridDim.x * 256;
  for (; i < n; i += stride) out[i] = (__bf16)in[i];
}

// ---------------------------------------------------------------------------
// f32 [R,C] -> bf16 [C,R] transpose+convert (weights, done once).
// ---------------------------------------------------------------------------
__global__ __launch_bounds__(256) void transpose_f32_bf16(const float* __restrict__ in,
                                                          __bf16* __restrict__ out,
                                                          int R, int C) {
  __shared__ __bf16 t[32][33];
  const int c0 = blockIdx.x * 32, r0 = blockIdx.y * 32;
  const int x = threadIdx.x, y = threadIdx.y;
#pragma unroll
  for (int j = 0; j < 4; ++j)
    t[y + 8 * j][x] = (__bf16)in[(size_t)(r0 + y + 8 * j) * C + c0 + x];
  __syncthreads();
#pragma unroll
  for (int j = 0; j < 4; ++j)
    out[(size_t)(c0 + y + 8 * j) * R + r0 + x] = t[x][y + 8 * j];
}

// ---------------------------------------------------------------------------
// V [B,S,H*D] -> Vt [(b*H+h), D, S] per-head transpose (bf16 -> bf16)
// ---------------------------------------------------------------------------
__global__ __launch_bounds__(256) void transpose_v(const __bf16* __restrict__ Vb,
                                                   __bf16* __restrict__ Vt) {
  __shared__ __bf16 t[32][33];
  const int z = blockIdx.z;          // b*H + h
  const int b = z >> 3, h = z & 7;
  const int n0 = blockIdx.x * 32;    // d index
  const int k0 = blockIdx.y * 32;    // seq index
  const int x = threadIdx.x, y = threadIdx.y;
  const __bf16* src = Vb + ((size_t)(b * S_ + k0)) * HD_ + h * D_ + n0;
#pragma unroll
  for (int j = 0; j < 4; ++j) t[y + 8 * j][x] = src[(size_t)(y + 8 * j) * HD_ + x];
  __syncthreads();
  __bf16* dst = Vt + ((size_t)z * D_ + n0) * S_ + k0;
#pragma unroll
  for (int j = 0; j < 4; ++j) dst[(size_t)(y + 8 * j) * S_ + x] = t[x][y + 8 * j];
}

// ---------------------------------------------------------------------------
// LDS tile loaders: 256 threads stage a 128x32 bf16 tile.
// ---------------------------------------------------------------------------
__device__ __forceinline__ void load_tile_async_bf16(__bf16* __restrict__ dst,
                                                     const __bf16* __restrict__ src,
                                                     int ld, int tid) {
  const int row = tid >> 1;
  const int seg = (tid & 1) * 16;
  async_copy32B(dst + row * LDT + seg, src + (size_t)row * ld + seg);
}

// f32 rows-major source converted to bf16 on the way to LDS (attn tiles).
__device__ __forceinline__ void load_tile_rows_f32(__bf16* __restrict__ dst,
                                                   const float* __restrict__ src,
                                                   int ld, int tid) {
  const int row = tid >> 1;
  const int seg = (tid & 1) * 16;
  const float* s = src + (size_t)row * ld + seg;
  __builtin_prefetch(s + BK, 0, 0);      // global_prefetch_b8 next k-tile
  __bf16 tmp[16];
#pragma unroll
  for (int j = 0; j < 16; ++j) tmp[j] = (__bf16)s[j];
  *(uint4*)(dst + row * LDT + seg)     = *(uint4*)(tmp);
  *(uint4*)(dst + row * LDT + seg + 8) = *(uint4*)(tmp + 8);
}

// ---------------------------------------------------------------------------
// WMMA fragment gathers (ISA 16-bit A 16x32 / B 32x16 wave32 layouts)
// ---------------------------------------------------------------------------
__device__ __forceinline__ v16bf frag_a(const __bf16* __restrict__ As, int row, int lane) {
  const int lh = lane >> 4, lm = lane & 15;
  const __bf16* p = As + (row + lm) * LDT;
  v8bf lo = *(const v8bf*)(p + lh * 8);        // K = lh*8 .. +7
  v8bf hi = *(const v8bf*)(p + 16 + lh * 8);   // K = 16+lh*8 .. +7
  v16bf r;
#pragma unroll
  for (int i = 0; i < 8; ++i) { r[i] = lo[i]; r[i + 8] = hi[i]; }
  return r;
}

__device__ __forceinline__ v16bf frag_b(const __bf16* __restrict__ Bs, int col, int lane) {
  const int lh = lane >> 4, lm = lane & 15;
  const __bf16* p = Bs + (col + lm) * LDT;
  v8bf lo = *(const v8bf*)(p + lh * 8);        // K = lh*8 .. +7
  v8bf hi = *(const v8bf*)(p + 16 + lh * 8);   // K = 16+lh*8 .. +7
  v16bf r;
#pragma unroll
  for (int i = 0; i < 8; ++i) { r[i] = lo[i]; r[i + 8] = hi[i]; }
  return r;
}

// ---------------------------------------------------------------------------
// Unified tiled GEMM kernel. B operand is always N-major [N, K].
// ---------------------------------------------------------------------------
enum GemmKind { K_PROJ, K_SCORES, K_CTX, K_OUT };

template <GemmKind KIND> struct GT;
template <> struct GT<K_PROJ>   { static constexpr int Kd = D_;  static constexpr int lda = D_;  static constexpr int ldb = D_;  static constexpr int ldc = HD_; };
template <> struct GT<K_SCORES> { static constexpr int Kd = D_;  static constexpr int lda = HD_; static constexpr int ldb = HD_; static constexpr int ldc = S_;  };
template <> struct GT<K_CTX>    { static constexpr int Kd = S_;  static constexpr int lda = S_;  static constexpr int ldb = S_;  static constexpr int ldc = HD_; };
template <> struct GT<K_OUT>    { static constexpr int Kd = HD_; static constexpr int lda = HD_; static constexpr int ldb = HD_; static constexpr int ldc = D_;  };

template <GemmKind KIND>
__global__ __launch_bounds__(256) void gemm_wmma(const void* __restrict__ aBase,
                                                 const void* __restrict__ bBase,
                                                 const float* __restrict__ bias,
                                                 void* __restrict__ cBase) {
  constexpr int Kd  = GT<KIND>::Kd;
  constexpr int lda = GT<KIND>::lda;
  constexpr int ldb = GT<KIND>::ldb;
  constexpr int ldc = GT<KIND>::ldc;
  constexpr int NT  = Kd / BK;          // k-tiles (>= 16)

  __shared__ __bf16 As[2][BM * LDT];    // double-buffered: DMA overlaps WMMA
  __shared__ __bf16 Bs[2][BN * LDT];

  const int tid  = threadIdx.x;
  const int lane = tid & 31;
  const int wave = tid >> 5;
  const int wr   = (wave >> 2) * 64;    // wave row offset (2 wave-rows x 64)
  const int wc   = (wave & 3) * 32;     // wave col offset (4 wave-cols x 32)
  const int mBase = blockIdx.y * BM;
  const int nBase = blockIdx.x * BN;

  const int z  = blockIdx.z;            // h*B + b for attention kernels
  const int bb = z & (B_ - 1);
  const int hh = z >> 2;

  const __bf16* Ab = nullptr;
  const float*  Af = nullptr;
  const __bf16* Bb = nullptr;
  float*        Cf = nullptr;
  __bf16*       Cb = nullptr;

  if constexpr (KIND == K_SCORES) {
    const size_t off = (size_t)bb * S_ * HD_ + (size_t)hh * D_;
    Ab = (const __bf16*)aBase + off;            // Q_hb [S,512] stride 4096
    Bb = (const __bf16*)bBase + off;            // K_hb [S,512] stride 4096 (N-major)
    Cf = (float*)cBase + (size_t)z * S_ * S_;   // attn[z]
  } else if constexpr (KIND == K_CTX) {
    Af = (const float*)aBase + (size_t)z * S_ * S_;                 // attn[z] f32
    Bb = (const __bf16*)bBase + (size_t)(bb * H_ + hh) * D_ * S_;   // Vt_hb [512,1024]
    Cb = (__bf16*)cBase + (size_t)bb * S_ * HD_ + (size_t)hh * D_;  // ctx slot
  } else {
    Ab = (const __bf16*)aBase;
    Bb = (const __bf16*)bBase;                  // transposed weights, N-major
    if constexpr (KIND == K_PROJ) Cb = (__bf16*)cBase;
    else                          Cf = (float*)cBase;
  }

  // Stage k-tile kb into LDS buffer `buf`: TDM for A (one wave), async for B.
  auto stage = [&](int kb, int buf) {
    if constexpr (KIND == K_CTX) {
      load_tile_rows_f32(As[buf], Af + (size_t)mBase * lda + kb, lda, tid);
    } else {
      if (wave == 0)
        tdm_load_tile_a(Ab + (size_t)mBase * lda + kb, As[buf], lda);
    }
    load_tile_async_bf16(Bs[buf], Bb + (size_t)nBase * ldb + kb, ldb, tid);
  };

  v8f acc[4][2] = {};

  stage(0, 0);                          // prologue: fill buffer 0
  for (int it = 0; it < NT; ++it) {
    const int cur = it & 1;
    if (it + 1 < NT) {
      stage((it + 1) * BK, cur ^ 1);    // next tile's DMA runs during compute
      wait_async_le2();                 // oldest 2 asyncs (cur B tile) landed
      if constexpr (KIND != K_CTX) {
        if (wave == 0) __builtin_amdgcn_s_wait_tensorcnt(1);  // cur A tile landed
      }
    } else {
      wait_async_le0();
      if constexpr (KIND != K_CTX) {
        if (wave == 0) __builtin_amdgcn_s_wait_tensorcnt(0);
      }
    }
    __syncthreads();                    // cur buffer visible to all waves

    v16bf a[4], b[2];
#pragma unroll
    for (int i = 0; i < 4; ++i) a[i] = frag_a(As[cur], wr + 16 * i, lane);
#pragma unroll
    for (int j = 0; j < 2; ++j) b[j] = frag_b(Bs[cur], wc + 16 * j, lane);

#pragma unroll
    for (int i = 0; i < 4; ++i)
#pragma unroll
      for (int j = 0; j < 2; ++j)
        acc[i][j] = __builtin_amdgcn_wmma_f32_16x16x32_bf16(
            false, a[i], false, b[j], (short)0, acc[i][j], false, false);

    __syncthreads();                    // reads done before cur is refilled
  }

  // Epilogue: D layout row = lh*8 + i, col = lane&15
  const int lh = lane >> 4, lm = lane & 15;
#pragma unroll
  for (int tm = 0; tm < 4; ++tm)
#pragma unroll
    for (int tn = 0; tn < 2; ++tn)
#pragma unroll
      for (int i = 0; i < 8; ++i) {
        const int r = mBase + wr + tm * 16 + lh * 8 + i;
        const int c = nBase + wc + tn * 16 + lm;
        float v = acc[tm][tn][i];
        if constexpr (KIND == K_SCORES) v *= (1.0f / (float)D_);
        if constexpr (KIND == K_PROJ || KIND == K_OUT) v += bias[c];
        if constexpr (KIND == K_PROJ || KIND == K_CTX) Cb[(size_t)r * ldc + c] = (__bf16)v;
        else                                           Cf[(size_t)r * ldc + c] = v;
      }
}

// ---------------------------------------------------------------------------
// Row softmax over 1024-wide rows, in place. One 256-thread block per row.
// ---------------------------------------------------------------------------
__global__ __launch_bounds__(256) void softmax_rows(float* __restrict__ attn) {
  __shared__ float red[256];
  float* p = attn + (size_t)blockIdx.x * S_;
  const int t = threadIdx.x;

  float4 v = ((float4*)p)[t];
  float mx = fmaxf(fmaxf(v.x, v.y), fmaxf(v.z, v.w));
  red[t] = mx;
  __syncthreads();
#pragma unroll
  for (int s = 128; s > 0; s >>= 1) {
    if (t < s) red[t] = fmaxf(red[t], red[t + s]);
    __syncthreads();
  }
  mx = red[0];
  __syncthreads();

  v.x = __expf(v.x - mx);
  v.y = __expf(v.y - mx);
  v.z = __expf(v.z - mx);
  v.w = __expf(v.w - mx);
  red[t] = v.x + v.y + v.z + v.w;
  __syncthreads();
#pragma unroll
  for (int s = 128; s > 0; s >>= 1) {
    if (t < s) red[t] += red[t + s];
    __syncthreads();
  }
  const float inv = 1.0f / red[0];
  v.x *= inv; v.y *= inv; v.z *= inv; v.w *= inv;
  ((float4*)p)[t] = v;
}

// ---------------------------------------------------------------------------
// Host-side orchestration
// ---------------------------------------------------------------------------
extern "C" void kernel_launch(void* const* d_in, const int* in_sizes, int n_in,
                              void* d_out, int out_size, void* d_ws, size_t ws_size,
                              hipStream_t stream) {
  const float* x  = (const float*)d_in[0];
  const float* Wq = (const float*)d_in[1];
  const float* bq = (const float*)d_in[2];
  const float* Wk = (const float*)d_in[3];
  const float* bk = (const float*)d_in[4];
  const float* Wv = (const float*)d_in[5];
  const float* bv = (const float*)d_in[6];
  const float* Wo = (const float*)d_in[7];
  const float* bo = (const float*)d_in[8];

  // Workspace layout (bf16), ~180 MB total
  __bf16* xb  = (__bf16*)d_ws;                    // 4096 x  512
  __bf16* Wqt = xb  + (size_t)M_ * D_;            // 4096 x  512  (= Wq^T)
  __bf16* Wkt = Wqt + (size_t)HD_ * D_;
  __bf16* Wvt = Wkt + (size_t)HD_ * D_;
  __bf16* Wot = Wvt + (size_t)HD_ * D_;           //  512 x 4096  (= Wo^T)
  __bf16* Qb  = Wot + (size_t)D_ * HD_;           // 4096 x 4096 [B,S,H*D]
  __bf16* Kb  = Qb  + (size_t)M_ * HD_;
  __bf16* Vb  = Kb  + (size_t)M_ * HD_;
  __bf16* Vt  = Vb  + (size_t)M_ * HD_;           // [(b*H+h), 512, 1024]
  __bf16* Cxb = Vt  + (size_t)M_ * HD_;           // ctx 4096 x 4096

  float* out  = (float*)d_out;                    // [B,S,D]   = 2,097,152 floats
  float* attn = out + (size_t)M_ * D_;            // [H*B,S,S] = 33,554,432 floats

  const dim3 tb(32, 8);
  cvt_f32_bf16<<<2048, 256, 0, stream>>>(x, xb, M_ * D_);
  transpose_f32_bf16<<<dim3(HD_ / 32, D_ / 32), tb, 0, stream>>>(Wq, Wqt, D_, HD_);
  transpose_f32_bf16<<<dim3(HD_ / 32, D_ / 32), tb, 0, stream>>>(Wk, Wkt, D_, HD_);
  transpose_f32_bf16<<<dim3(HD_ / 32, D_ / 32), tb, 0, stream>>>(Wv, Wvt, D_, HD_);
  transpose_f32_bf16<<<dim3(D_ / 32, HD_ / 32), tb, 0, stream>>>(Wo, Wot, HD_, D_);

  const dim3 blk(256);
  // QKV projections: [4096,512] @ [512,4096] + bias -> bf16
  gemm_wmma<K_PROJ><<<dim3(HD_ / BN, M_ / BM), blk, 0, stream>>>(xb, Wqt, bq, Qb);
  gemm_wmma<K_PROJ><<<dim3(HD_ / BN, M_ / BM), blk, 0, stream>>>(xb, Wkt, bk, Kb);
  gemm_wmma<K_PROJ><<<dim3(HD_ / BN, M_ / BM), blk, 0, stream>>>(xb, Wvt, bv, Vb);

  // Per-head V transpose for N-major ctx GEMM
  transpose_v<<<dim3(D_ / 32, S_ / 32, B_ * H_), tb, 0, stream>>>(Vb, Vt);

  // scores = Q K^T / 512 per (h,b) -> f32 into attn region of d_out
  gemm_wmma<K_SCORES><<<dim3(S_ / BN, S_ / BM, H_ * B_), blk, 0, stream>>>(Qb, Kb, nullptr, attn);

  // softmax rows in place
  softmax_rows<<<H_ * B_ * S_, 256, 0, stream>>>(attn);

  // ctx = attn @ V per (h,b) -> bf16 [B,S,H*D]
  gemm_wmma<K_CTX><<<dim3(D_ / BN, S_ / BM, H_ * B_), blk, 0, stream>>>(attn, Vt, nullptr, Cxb);

  // out = ctx @ Wo + bo -> f32
  gemm_wmma<K_OUT><<<dim3(D_ / BN, M_ / BM), blk, 0, stream>>>(Cxb, Wot, bo, out);
}
